// MultiHeadAttention_14216341749984
// MI455X (gfx1250) — matrix operand order repaired and checked by reading, not compiled, not run on hardware
//
#include <hip/hip_runtime.h>
#include <hip/hip_bf16.h>

#define BB 2
#define SS 2048
#define DD 512
#define HH 8
#define DH 64
#define QB 64                // queries per workgroup (4 waves x 16-row strips)
#define PROWS (QB + 1)       // skew needs rows i0..i0+QB

typedef __attribute__((ext_vector_type(16))) _Float16 v16h;
typedef __attribute__((ext_vector_type(8)))  float    v8f;

union FragH { v16h v; unsigned u[8]; _Float16 h[16]; };

// ISA 7.12.2: 16-bit A 16x32 per-lane K index for VGPR v, lane-half hf
__device__ __forceinline__ int a_kb(int v, int hf) { return ((v >> 2) << 4) + (hf << 3) + ((v & 3) << 1); }
// 16-bit B 32x16 per-lane K index
__device__ __forceinline__ int b_kb(int v, int hf) { return ((v >> 2) << 4) + (hf << 3) + ((v & 3) << 1); }

__device__ __forceinline__ v8f wmma_f16(const FragH& a, const FragH& b, v8f c) {
  return __builtin_amdgcn_wmma_f32_16x16x32_f16(false, a.v, false, b.v, (short)0, c, false, false);
}

// ---------------------------------------------------------------------------
// Tiled WMMA GEMM: out = A[MxK] * W[KxN] + bias (+ optional extra biases,
// dual output for qc/qp). A is f32 or f16; output f16 (workspace) or f32.
// Block: 256 threads = 8 waves; block tile 128x128; wave tile 64x32.
// ---------------------------------------------------------------------------
template <typename TA, typename TO>
__global__ __launch_bounds__(256)
void proj_gemm(const TA* __restrict__ A, const float* __restrict__ W,
               const float* __restrict__ bias,
               const float* __restrict__ eb1, TO* __restrict__ out1,
               const float* __restrict__ eb2, TO* __restrict__ out2,
               int M, int N, int K)
{
  __shared__ _Float16 As[128 * 40];   // A tile  (row-major, stride 40)
  __shared__ _Float16 Bs[128 * 40];   // W tile, transposed: Bs[n][k]

  const int tid = threadIdx.x;
  const int hf  = (tid >> 4) & 1;
  const int ln  = tid & 15;
  const int wave = tid >> 5;
  const int wy = wave >> 2, wx = wave & 3;
  const int r0 = blockIdx.x * 128, n0 = blockIdx.y * 128;

  v8f acc[4][2];
  #pragma unroll
  for (int i = 0; i < 4; ++i) {
    #pragma unroll
    for (int j = 0; j < 2; ++j) acc[i][j] = {};
  }

  for (int k0 = 0; k0 < K; k0 += 32) {
    {
      const int row = tid >> 1, cb = (tid & 1) * 16;
      const TA* src = A + (size_t)(r0 + row) * K + k0 + cb;
      #pragma unroll
      for (int c = 0; c < 16; ++c) As[row * 40 + cb + c] = (_Float16)(float)src[c];
      // W tile transposed into Bs[n][k]
      #pragma unroll
      for (int c = 0; c < 16; ++c)
        Bs[row * 40 + cb + c] = (_Float16)W[(size_t)(k0 + cb + c) * N + n0 + row];
    }
    __syncthreads();

    FragH af[4], bf[2];
    #pragma unroll
    for (int rt = 0; rt < 4; ++rt) {
      const _Float16* p = As + (wy * 64 + rt * 16 + ln) * 40;
      #pragma unroll
      for (int v = 0; v < 8; ++v) af[rt].u[v] = *(const unsigned*)(p + a_kb(v, hf));
    }
    #pragma unroll
    for (int nt = 0; nt < 2; ++nt) {
      const _Float16* p = Bs + (wx * 32 + nt * 16 + ln) * 40;
      #pragma unroll
      for (int v = 0; v < 8; ++v) bf[nt].u[v] = *(const unsigned*)(p + b_kb(v, hf));
    }
    #pragma unroll
    for (int rt = 0; rt < 4; ++rt) {
      #pragma unroll
      for (int nt = 0; nt < 2; ++nt)
        acc[rt][nt] = wmma_f16(af[rt], bf[nt], acc[rt][nt]);
    }
    __syncthreads();
  }

  #pragma unroll
  for (int rt = 0; rt < 4; ++rt) {
    #pragma unroll
    for (int nt = 0; nt < 2; ++nt) {
      const int col = n0 + wx * 32 + nt * 16 + ln;
      const float bcol = bias[col];
      const float e1 = eb1 ? eb1[col] : 0.f;
      #pragma unroll
      for (int r = 0; r < 8; ++r) {
        const int row = r0 + wy * 64 + rt * 16 + (hf << 3) + r;
        const float val = acc[rt][nt][r] + bcol;
        out1[(size_t)row * N + col] = (TO)(val + e1);
        if (out2) out2[(size_t)row * N + col] = (TO)(val + eb2[col]);
      }
    }
  }
}

// ---------------------------------------------------------------------------
// Fused Transformer-XL relative attention (flash-style, 64-query blocks).
// Block = 128 threads = 4 waves (one per SIMD32); wave owns a 16-row strip.
// Phase 1: P[i0..i0+64][0..S) = Qp * Ph^T into LDS (f16) via WMMA.
// Phase 2: flash loop over 64-key blocks: content WMMA + skew gather from
//          P-LDS + online softmax + attn*V WMMA.  LDS total ~294 KB < 320 KB.
// ---------------------------------------------------------------------------
__global__ __launch_bounds__(128)
void relattn(const _Float16* __restrict__ qc, const _Float16* __restrict__ qp,
             const _Float16* __restrict__ kh, const _Float16* __restrict__ vh,
             const _Float16* __restrict__ ph, _Float16* __restrict__ ctx)
{
  extern __shared__ _Float16 smem[];
  _Float16* Plds  = smem;                       // PROWS * SS
  _Float16* Klds  = Plds + PROWS * SS;          // 64 * 72  (rows of K = B^T)
  _Float16* Vtlds = Klds + 64 * 72;             // 64 * 72  (V transposed: [d][t])
  _Float16* Atlds = Vtlds + 64 * 72;            // 4 waves * 16 * 72 (attn tiles)

  const int tid = threadIdx.x;
  const int hf  = (tid >> 4) & 1;
  const int ln  = tid & 15;
  const int wave = tid >> 5;                    // 0..3
  const int i0 = blockIdx.x * QB;
  const int h  = blockIdx.y, b = blockIdx.z;

  const size_t base = (size_t)b * SS * DD + (size_t)h * DH;
  const _Float16* qcB = qc + base;
  const _Float16* qpB = qp + base;
  const _Float16* khB = kh + base;
  const _Float16* vhB = vh + base;
  const _Float16* phB = ph + base;
  _Float16* ctxB = ctx + base;

  // ---- Phase 1: relative-position scores, rows i0..i0+64, all 2048 cols ----
  for (int rt = 0; rt < 5; ++rt) {              // 80 rows computed, 65 stored
    FragH aq[2];
    int rowq = i0 + rt * 16 + ln;
    if (rowq > SS - 1) rowq = SS - 1;           // clamp (extra rows unused)
    #pragma unroll
    for (int ks = 0; ks < 2; ++ks) {
      #pragma unroll
      for (int v = 0; v < 8; ++v)
        aq[ks].u[v] = *(const unsigned*)(qpB + (size_t)rowq * DD + ks * 32 + a_kb(v, hf));
    }
    for (int ct = wave * 32; ct < wave * 32 + 32; ++ct) {   // waves split columns
      v8f acc = {};
      #pragma unroll
      for (int ks = 0; ks < 2; ++ks) {
        FragH bp_;
        const int t = ct * 16 + ln;             // ph row t is B^T row directly
        #pragma unroll
        for (int v = 0; v < 8; ++v)
          bp_.u[v] = *(const unsigned*)(phB + (size_t)t * DD + ks * 32 + b_kb(v, hf));
        acc = wmma_f16(aq[ks], bp_, acc);
      }
      #pragma unroll
      for (int r = 0; r < 8; ++r) {
        const int ml = rt * 16 + (hf << 3) + r;
        if (ml < PROWS) Plds[ml * SS + ct * 16 + ln] = (_Float16)acc[r];
      }
    }
  }
  __syncthreads();

  // ---- Phase 2: flash attention over 64-key blocks ----
  FragH qa[2];
  {
    const int rowq = i0 + wave * 16 + ln;
    #pragma unroll
    for (int ks = 0; ks < 2; ++ks) {
      #pragma unroll
      for (int v = 0; v < 8; ++v)
        qa[ks].u[v] = *(const unsigned*)(qcB + (size_t)rowq * DD + ks * 32 + a_kb(v, hf));
    }
  }

  float mrow[8], lrow[8];
  v8f o[4];
  #pragma unroll
  for (int r = 0; r < 8; ++r) { mrow[r] = -1e30f; lrow[r] = 0.f; }
  #pragma unroll
  for (int dt = 0; dt < 4; ++dt) o[dt] = {};

  _Float16* atw = Atlds + wave * 16 * 72;

  for (int jb = 0; jb < SS / 64; ++jb) {
    const int j0 = jb * 64;
    {   // stage K rows and V transposed (128 threads -> 2 threads per row)
      const int t  = tid >> 1;
      const int cb = (tid & 1) * 32;            // half-row of 32 elements
      const uint4* ksrc = (const uint4*)(khB + (size_t)(j0 + t) * DD + cb);
      uint4* kdst = (uint4*)(Klds + t * 72 + cb);
      #pragma unroll
      for (int c = 0; c < 4; ++c) kdst[c] = ksrc[c];
      __align__(16) _Float16 vr[32];
      const uint4* vsrc = (const uint4*)(vhB + (size_t)(j0 + t) * DD + cb);
      #pragma unroll
      for (int c = 0; c < 4; ++c) *(uint4*)&vr[c * 8] = vsrc[c];
      #pragma unroll
      for (int d = 0; d < 32; ++d) Vtlds[(cb + d) * 72 + t] = vr[d];
      if (j0 + 64 < SS) {
        __builtin_prefetch(khB + (size_t)(j0 + 64 + t) * DD + cb, 0, 1);
        __builtin_prefetch(vhB + (size_t)(j0 + 64 + t) * DD + cb, 0, 1);
      }
    }
    __syncthreads();

    // content scores (16x64 strip) + skewed pos gather + scale
    float sv[4][8];
    #pragma unroll
    for (int nt = 0; nt < 4; ++nt) {
      v8f acc = {};
      #pragma unroll
      for (int ks = 0; ks < 2; ++ks) {
        FragH bk_;
        const _Float16* p = Klds + (nt * 16 + ln) * 72 + ks * 32;
        #pragma unroll
        for (int v = 0; v < 8; ++v) bk_.u[v] = *(const unsigned*)(p + b_kb(v, hf));
        acc = wmma_f16(qa[ks], bk_, acc);
      }
      #pragma unroll
      for (int r = 0; r < 8; ++r) {
        const int i = i0 + wave * 16 + (hf << 3) + r;
        const int j = j0 + nt * 16 + ln;
        float pv;
        if (j == i + 1)      pv = 0.f;
        else if (j <= i)     pv = (float)Plds[(i - i0) * SS + (SS - 1 - i + j)];
        else                 pv = (float)Plds[(i + 1 - i0) * SS + (j - i - 2)];
        sv[nt][r] = (acc[r] + pv) * 0.125f;     // 1/sqrt(Dh)
      }
    }

    // online softmax (rows live in 16-lane groups: shfl_xor 1,2,4,8)
    #pragma unroll
    for (int r = 0; r < 8; ++r) {
      float tm = sv[0][r];
      #pragma unroll
      for (int nt = 1; nt < 4; ++nt) tm = fmaxf(tm, sv[nt][r]);
      #pragma unroll
      for (int off = 1; off < 16; off <<= 1) tm = fmaxf(tm, __shfl_xor(tm, off, 32));
      const float mn = fmaxf(mrow[r], tm);
      const float corr = __expf(mrow[r] - mn);
      float rs = 0.f;
      #pragma unroll
      for (int nt = 0; nt < 4; ++nt) {
        const float pe = __expf(sv[nt][r] - mn);
        atw[((hf << 3) + r) * 72 + nt * 16 + ln] = (_Float16)pe;
        rs += pe;
      }
      #pragma unroll
      for (int off = 1; off < 16; off <<= 1) rs += __shfl_xor(rs, off, 32);
      lrow[r] = lrow[r] * corr + rs;
      mrow[r] = mn;
      #pragma unroll
      for (int dt = 0; dt < 4; ++dt) o[dt][r] *= corr;
    }

    // O(16x64) += attn(16x64keys) * V(64keys x 64d)
    #pragma unroll
    for (int ks = 0; ks < 2; ++ks) {
      FragH pa;
      const _Float16* p = atw + ln * 72 + ks * 32;
      #pragma unroll
      for (int v = 0; v < 8; ++v) pa.u[v] = *(const unsigned*)(p + a_kb(v, hf));
      #pragma unroll
      for (int dt = 0; dt < 4; ++dt) {
        FragH vb;
        const _Float16* q_ = Vtlds + (dt * 16 + ln) * 72 + ks * 32;
        #pragma unroll
        for (int v = 0; v < 8; ++v) vb.u[v] = *(const unsigned*)(q_ + b_kb(v, hf));
        o[dt] = wmma_f16(pa, vb, o[dt]);
      }
    }
    __syncthreads();
  }

  // normalize and write ctx (f16 workspace, [b*S+i][h*Dh+d] layout)
  #pragma unroll
  for (int r = 0; r < 8; ++r) {
    const int i = i0 + wave * 16 + (hf << 3) + r;
    const float inv = 1.f / lrow[r];
    #pragma unroll
    for (int dt = 0; dt < 4; ++dt)
      ctxB[(size_t)i * DD + dt * 16 + ln] = (_Float16)(o[dt][r] * inv);
  }
}

// ---------------------------------------------------------------------------
extern "C" void kernel_launch(void* const* d_in, const int* in_sizes, int n_in,
                              void* d_out, int out_size, void* d_ws, size_t ws_size,
                              hipStream_t stream)
{
  (void)in_sizes; (void)n_in; (void)out_size; (void)ws_size;

  const float* q   = (const float*)d_in[0];
  const float* k   = (const float*)d_in[1];
  const float* v   = (const float*)d_in[2];
  const float* pe  = (const float*)d_in[3];
  const float* Wq  = (const float*)d_in[4];
  const float* bq  = (const float*)d_in[5];
  const float* Wk  = (const float*)d_in[6];
  const float* bk  = (const float*)d_in[7];
  const float* Wv  = (const float*)d_in[8];
  const float* bv  = (const float*)d_in[9];
  const float* Wp  = (const float*)d_in[10];
  const float* bp  = (const float*)d_in[11];
  const float* Wo  = (const float*)d_in[12];
  const float* bo  = (const float*)d_in[13];
  const float* cb  = (const float*)d_in[14];   // content bias  [H*Dh]
  const float* pb_ = (const float*)d_in[15];   // position bias [H*Dh]

  const int M = BB * SS;                       // 4096 rows
  const size_t SZ = (size_t)M * DD;            // elements per projection

  _Float16* wsh  = (_Float16*)d_ws;
  _Float16* qcW  = wsh;
  _Float16* qpW  = wsh + SZ;
  _Float16* khW  = wsh + 2 * SZ;
  _Float16* vhW  = wsh + 3 * SZ;
  _Float16* phW  = wsh + 4 * SZ;
  _Float16* ctxW = wsh + 5 * SZ;

  dim3 pgrid(M / 128, DD / 128);
  proj_gemm<float, _Float16><<<pgrid, 256, 0, stream>>>(q,  Wq, bq, cb,      qcW, pb_,     qpW, M, DD, DD);
  proj_gemm<float, _Float16><<<pgrid, 256, 0, stream>>>(k,  Wk, bk, nullptr, khW, nullptr, (_Float16*)nullptr, M, DD, DD);
  proj_gemm<float, _Float16><<<pgrid, 256, 0, stream>>>(v,  Wv, bv, nullptr, vhW, nullptr, (_Float16*)nullptr, M, DD, DD);
  proj_gemm<float, _Float16><<<pgrid, 256, 0, stream>>>(pe, Wp, bp, nullptr, phW, nullptr, (_Float16*)nullptr, M, DD, DD);

  const size_t smem = (size_t)(PROWS * SS + 2 * 64 * 72 + 4 * 16 * 72) * sizeof(_Float16);
  relattn<<<dim3(SS / QB, HH, BB), 128, smem, stream>>>(qcW, qpW, khW, vhW, phW, ctxW);

  proj_gemm<_Float16, float><<<pgrid, 256, 0, stream>>>(ctxW, Wo, bo, nullptr, (float*)d_out, nullptr, (float*)nullptr, M, DD, DD);
}
